// SAGE_30640296689761
// MI455X (gfx1250) — matrix operand order repaired and checked by reading, not compiled, not run on hardware
//
#include <hip/hip_runtime.h>
#include <hip/hip_bf16.h>

#define HID 128
#define LAT 64
#define BN_EPS 1e-5f
#define LRELU 0.2f

typedef _Float16 v16h __attribute__((ext_vector_type(16)));
typedef float    v8f  __attribute__((ext_vector_type(8)));

// ---------------- utility kernels ----------------

__global__ void zero_kernel(float* __restrict__ p, size_t n) {
    size_t i = (size_t)blockIdx.x * blockDim.x + threadIdx.x;
    size_t stride = (size_t)gridDim.x * blockDim.x;
    for (; i < n; i += stride) p[i] = 0.0f;
}

__global__ void deg_kernel(const int* __restrict__ dst, float* __restrict__ deg, int E) {
    int e = blockIdx.x * blockDim.x + threadIdx.x;
    if (e < E) atomicAdd(&deg[dst[e]], 1.0f);
}

__global__ void invdeg_kernel(float* __restrict__ deg, int N) {
    int i = blockIdx.x * blockDim.x + threadIdx.x;
    if (i < N) deg[i] = 1.0f / fmaxf(deg[i], 1.0f);
}

// One wave per edge; each lane moves 4 features (float4 gather, 4 f32 atomics).
__global__ void scatter_kernel(const float* __restrict__ x,
                               const int* __restrict__ src,
                               const int* __restrict__ dst,
                               float* __restrict__ agg, int E) {
    int gid  = blockIdx.x * blockDim.x + threadIdx.x;
    int e    = gid >> 5;
    int lane = gid & 31;
    if (e >= E) return;
    int s = src[e];
    int d = dst[e];
    const float4 v = *(const float4*)(x + (size_t)s * HID + lane * 4);
    float* bp = agg + (size_t)d * HID + lane * 4;
    atomicAdd(bp + 0, v.x);
    atomicAdd(bp + 1, v.y);
    atomicAdd(bp + 2, v.z);
    atomicAdd(bp + 3, v.w);
}

// One wave per node: global_add_pool via atomics.
__global__ void pool_kernel(const float* __restrict__ x,
                            const int* __restrict__ batch,
                            float* __restrict__ pool, int N) {
    int gid  = blockIdx.x * blockDim.x + threadIdx.x;
    int n    = gid >> 5;
    int lane = gid & 31;
    if (n >= N) return;
    int g = batch[n];
    const float4 v = *(const float4*)(x + (size_t)n * HID + lane * 4);
    float* bp = pool + (size_t)g * HID + lane * 4;
    atomicAdd(bp + 0, v.x);
    atomicAdd(bp + 1, v.y);
    atomicAdd(bp + 2, v.z);
    atomicAdd(bp + 3, v.w);
}

__global__ void bn_kernel(const float* __restrict__ in,
                          const float* __restrict__ g,
                          const float* __restrict__ b,
                          float* __restrict__ out, int total) {
    int i = blockIdx.x * blockDim.x + threadIdx.x;
    if (i >= total) return;
    int f = i & (HID - 1);
    out[i] = in[i] * (g[f] * rsqrtf(1.0f + BN_EPS)) + b[f];
}

// ---------------- weight pre-pack into WMMA B-fragment layout ----------------
// W is [128 (out n)][128 (in k)] row-major. B fragment, contiguous k order (same as A):
//   lane = (n&15) + 16*((K>>4)&1), half h = K&15, chunk c = K>>5, ntile = n>>4.
// Packed layout: out[((ntile*4 + c)*32 + lane)*16 + h], f16.
__global__ void pack_w_kernel(const float* __restrict__ W, _Float16* __restrict__ out) {
    int t = blockIdx.x * blockDim.x + threadIdx.x;   // 128 rows * 32 float4 = 4096
    if (t >= HID * 32) return;
    int n  = t >> 5;
    int K0 = (t & 31) * 4;
    const float4 v = *(const float4*)(W + (size_t)n * HID + K0);
    int ntile = n >> 4;
    int c     = K0 >> 5;
    int lane  = (n & 15) + (((K0 >> 4) & 1) << 4);
    int h0    = K0 & 15;
    _Float16* dp = out + (((size_t)(ntile * 4 + c) * 32 + lane) * 16 + h0);
    dp[0] = (_Float16)v.x; dp[1] = (_Float16)v.y;
    dp[2] = (_Float16)v.z; dp[3] = (_Float16)v.w;
}

// ---------------- SAGEConv layer: relu(agg*invdeg @ Wl^T + x @ Wr^T + b) ----------------
// Block = 256 threads = 8 waves; block computes 16 rows x 128 cols; wave owns a 16-col tile.
// A/X panels staged in LDS directly as f16 A fragments, contiguous k order (same as B):
//   AF[m][c][lane][h] = S_m[lane&15][32*c + 16*(lane>>4) + h]
__global__ void __launch_bounds__(256)
conv_kernel(const float* __restrict__ xin, const float* __restrict__ agg,
            const float* __restrict__ inv_deg,
            const _Float16* __restrict__ WPl, const _Float16* __restrict__ WPr,
            const float* __restrict__ bias,
            float* __restrict__ xout, int Nnodes) {
    __shared__ __align__(32) _Float16 AF[2][4][32][16];   // 8 KB

    const int tid  = threadIdx.x;
    const int row0 = blockIdx.x * 16;

    // Prefetch next tile's streaming panels (global_prefetch_b8).
    if (tid < 16) {
        int nr = row0 + 16 + tid;
        if (nr < Nnodes) {
            __builtin_prefetch(xin + (size_t)nr * HID, 0, 1);
            __builtin_prefetch(agg + (size_t)nr * HID, 0, 1);
        }
    }

    // Stage panels (m=0: agg*invdeg, m=1: x) as f16 fragments. 1024 float4 loads / block.
    for (int t = tid; t < 2 * 16 * 32; t += 256) {
        int m    = t >> 9;
        int rest = t & 511;
        int r    = rest >> 5;
        int K0   = (rest & 31) * 4;
        int gr   = row0 + r;
        float4 v = make_float4(0.f, 0.f, 0.f, 0.f);
        if (gr < Nnodes) {
            const float* sp = (m ? xin : agg) + (size_t)gr * HID + K0;
            v = *(const float4*)sp;
            if (m == 0) {
                float id = inv_deg[gr];
                v.x *= id; v.y *= id; v.z *= id; v.w *= id;
            }
        }
        int c      = K0 >> 5;
        int lane_t = r + (((K0 >> 4) & 1) << 4);
        int h0     = K0 & 15;
        _Float16* dp = &AF[m][c][lane_t][h0];
        dp[0] = (_Float16)v.x; dp[1] = (_Float16)v.y;
        dp[2] = (_Float16)v.z; dp[3] = (_Float16)v.w;
    }
    __syncthreads();

    const int lane = tid & 31;
    const int wave = tid >> 5;
    const int hi   = lane >> 4;
    const int ncol = wave * 16 + (lane & 15);

    // C init = bias broadcast across the 8 row-slots this lane owns.
    v8f acc;
    float bv = bias[ncol];
#pragma unroll
    for (int i = 0; i < 8; ++i) acc[i] = bv;

#pragma unroll
    for (int pass = 0; pass < 2; ++pass) {
        const _Float16* WP = pass ? WPr : WPl;
#pragma unroll
        for (int c = 0; c < 4; ++c) {
            const v16h a = *(const v16h*)&AF[pass][c][lane][0];
            const v16h b = *(const v16h*)(WP + (((size_t)(wave * 4 + c) * 32 + lane) * 16));
            acc = __builtin_amdgcn_wmma_f32_16x16x32_f16(
                false, a, false, b, (short)0, acc, false, false);
        }
    }

    // Store with fused ReLU. C layout: VGPR v -> row (v + 8*hi), col = ncol.
#pragma unroll
    for (int v = 0; v < 8; ++v) {
        int gr = row0 + v + (hi << 3);
        if (gr < Nnodes) {
            float val = acc[v];
            xout[(size_t)gr * HID + ncol] = val > 0.0f ? val : 0.0f;
        }
    }
}

// ---------------- MLP layer: out = [bn](leakyrelu(in @ W^T + b)) ----------------
__global__ void __launch_bounds__(128)
mlp_kernel(const float* __restrict__ in, const float* __restrict__ W,
           const float* __restrict__ bias, float* __restrict__ out,
           int Nout, int use_bn,
           const float* __restrict__ g, const float* __restrict__ be) {
    __shared__ float sh[HID];
    const int row = blockIdx.x;
    const int t = threadIdx.x;
    sh[t] = in[(size_t)row * HID + t];
    __syncthreads();
    if (t < Nout) {
        float acc = bias[t];
        const float* wr = W + (size_t)t * HID;
#pragma unroll 8
        for (int k = 0; k < HID; ++k) acc += sh[k] * wr[k];
        float h = acc > 0.0f ? acc : LRELU * acc;
        if (use_bn) h = h * (g[t] * rsqrtf(1.0f + BN_EPS)) + be[t];
        out[(size_t)row * Nout + t] = h;
    }
}

// ---------------- driver ----------------

extern "C" void kernel_launch(void* const* d_in, const int* in_sizes, int n_in,
                              void* d_out, int out_size, void* d_ws, size_t ws_size,
                              hipStream_t stream) {
    const float* x_in  = (const float*)d_in[0];
    const int*   ei    = (const int*)d_in[1];
    const int*   batch = (const int*)d_in[2];
    const float* cWl   = (const float*)d_in[3];   // [3,128,128]
    const float* cWr   = (const float*)d_in[4];   // [3,128,128]
    const float* cb    = (const float*)d_in[5];   // [3,128]
    const float* bng   = (const float*)d_in[6];
    const float* bnb   = (const float*)d_in[7];
    const float* W1 = (const float*)d_in[8];  const float* b1 = (const float*)d_in[9];
    const float* g1 = (const float*)d_in[10]; const float* be1 = (const float*)d_in[11];
    const float* W2 = (const float*)d_in[12]; const float* b2 = (const float*)d_in[13];
    const float* g2 = (const float*)d_in[14]; const float* be2 = (const float*)d_in[15];
    const float* W3 = (const float*)d_in[16]; const float* b3 = (const float*)d_in[17];

    const int N = in_sizes[0] / HID;      // 100000
    const int E = in_sizes[1] / 2;        // 1600000
    const int G = out_size / LAT;         // 512

    // workspace partition (256B aligned)
    size_t off = 0;
    auto wsalloc = [&](size_t bytes) {
        char* p = (char*)d_ws + off;
        off += (bytes + 255) & ~(size_t)255;
        return (void*)p;
    };
    float* xA     = (float*)wsalloc((size_t)N * HID * sizeof(float));
    float* xB     = (float*)wsalloc((size_t)N * HID * sizeof(float));
    float* agg    = (float*)wsalloc((size_t)N * HID * sizeof(float));
    float* invdeg = (float*)wsalloc((size_t)N * sizeof(float));
    float* pooled = (float*)wsalloc((size_t)G * HID * sizeof(float));
    float* h0     = (float*)wsalloc((size_t)G * HID * sizeof(float));
    float* h1     = (float*)wsalloc((size_t)G * HID * sizeof(float));
    float* h2     = (float*)wsalloc((size_t)G * HID * sizeof(float));
    _Float16* WP  = (_Float16*)wsalloc((size_t)6 * HID * HID * sizeof(_Float16));
    (void)ws_size;

    const int* esrc = ei;
    const int* edst = ei + E;

    // 0) pre-pack 6 weight matrices into WMMA B-fragment f16 layout
    const int packBlocks = (HID * 32 + 255) / 256;
    for (int l = 0; l < 3; ++l) {
        pack_w_kernel<<<packBlocks, 256, 0, stream>>>(
            cWl + (size_t)l * HID * HID, WP + (size_t)(2 * l + 0) * HID * HID);
        pack_w_kernel<<<packBlocks, 256, 0, stream>>>(
            cWr + (size_t)l * HID * HID, WP + (size_t)(2 * l + 1) * HID * HID);
    }

    // 1) in-degree -> inv_deg (reuse buffer)
    zero_kernel<<<1024, 256, 0, stream>>>(invdeg, (size_t)N);
    deg_kernel<<<(E + 255) / 256, 256, 0, stream>>>(edst, invdeg, E);
    invdeg_kernel<<<(N + 255) / 256, 256, 0, stream>>>(invdeg, N);

    // 2) 3 SAGEConv layers (ping-pong xA/xB)
    const float* cur = x_in;
    float* bufs[2] = {xA, xB};
    const size_t scatterBlocks = ((size_t)E * 32 + 255) / 256;
    const int convBlocks = (N + 15) / 16;
    for (int l = 0; l < 3; ++l) {
        float* nxt = bufs[l & 1];
        zero_kernel<<<4096, 256, 0, stream>>>(agg, (size_t)N * HID);
        scatter_kernel<<<(unsigned)scatterBlocks, 256, 0, stream>>>(cur, esrc, edst, agg, E);
        conv_kernel<<<convBlocks, 256, 0, stream>>>(
            cur, agg, invdeg,
            WP + (size_t)(2 * l + 0) * HID * HID, WP + (size_t)(2 * l + 1) * HID * HID,
            cb + (size_t)l * HID, nxt, N);
        cur = nxt;
    }

    // 3) global_add_pool + BN
    zero_kernel<<<256, 256, 0, stream>>>(pooled, (size_t)G * HID);
    pool_kernel<<<(unsigned)(((size_t)N * 32 + 255) / 256), 256, 0, stream>>>(cur, batch, pooled, N);
    bn_kernel<<<(G * HID + 255) / 256, 256, 0, stream>>>(pooled, bng, bnb, h0, G * HID);

    // 4) MLP head
    mlp_kernel<<<G, 128, 0, stream>>>(h0, W1, b1, h1, HID, 1, g1, be1);
    mlp_kernel<<<G, 128, 0, stream>>>(h1, W2, b2, h2, HID, 1, g2, be2);
    mlp_kernel<<<G, 128, 0, stream>>>(h2, W3, b3, (float*)d_out, LAT, 0, nullptr, nullptr);
}
